// FlashChatglmAttention_39359080300888
// MI455X (gfx1250) — matrix operand order repaired and checked by reading, not compiled, not run on hardware
//
#include <hip/hip_runtime.h>

// ---- problem constants ----
#define HID      4096
#define NH       32
#define HS       128
#define S_LEN    1024
#define B_SZ     4
#define T_TOK    4096   // B*S
#define ROT_HALF 32

typedef __attribute__((ext_vector_type(16))) __bf16 v16bf;
typedef __attribute__((ext_vector_type(8)))  float  v8f;

union FragBF { v16bf v; unsigned u[8]; };

__device__ inline unsigned pack_bf16(float a, float b) {
    union { float f; unsigned u; } ua, ub;
    ua.f = a; ub.f = b;
    unsigned ra = (ua.u + 0x7FFFu + ((ua.u >> 16) & 1u)) >> 16;
    unsigned rb = (ub.u + 0x7FFFu + ((ub.u >> 16) & 1u)) >> 16;
    return (ra & 0xFFFFu) | (rb << 16);
}

__device__ inline v8f wmma_bf16(v16bf a, v16bf b, v8f c) {
    return __builtin_amdgcn_wmma_f32_16x16x32_bf16(
        false, a, false, b, (short)0, c, false, false);
}

// A-fragment 16x32 bf16 from LDS tile laid out [row][k], row stride STRIDE.
// ISA layout: lanes0-15 row=lane, v0..3 hold K{2v,2v+1}, v4..7 hold K{16+2v..};
// lanes16-31 same rows, K offset +8.
template<int STRIDE>
__device__ inline v16bf load_frag_a(const __bf16* lds, int row0, int k0, int lane) {
    int half = (lane >> 4) & 1;
    const __bf16* base = lds + (size_t)(row0 + (lane & 15)) * STRIDE + k0 + half * 8;
    FragBF f;
#pragma unroll
    for (int v = 0; v < 4; ++v)
        f.u[v] = *(const unsigned*)(base + 2 * v);
#pragma unroll
    for (int v = 0; v < 4; ++v)
        f.u[4 + v] = *(const unsigned*)(base + 16 + 2 * v);
    return f.v;
}

// B-fragment 32x16 bf16 from LDS tile laid out [col][k], col stride STRIDE.
// lanes0-15 col=lane, v0..3 hold K{2v,2v+1}, v4..7 hold K{16+2v..}; lanes16-31 K offset +8.
template<int STRIDE>
__device__ inline v16bf load_frag_b(const __bf16* lds, int col0, int k0, int lane) {
    int half = (lane >> 4) & 1;
    const __bf16* base = lds + (size_t)(col0 + (lane & 15)) * STRIDE + k0 + half * 8;
    FragBF f;
#pragma unroll
    for (int v = 0; v < 4; ++v)
        f.u[v] = *(const unsigned*)(base + 2 * v);
#pragma unroll
    for (int v = 0; v < 4; ++v)
        f.u[4 + v] = *(const unsigned*)(base + 16 + 2 * v);
    return f.v;
}

// ======================================================================
// GEMM: C[M,N] = A[M,K] @ B[K,N] (fp32 in, bf16 WMMA, fp32 accumulate)
// MODE 0: plain store to C.
// MODE 1: qkv epilogue: +bias, RoPE on first 64 dims of q/k heads,
//         scatter to q_out / k_out(kv cache) / v_out(kv cache).
// grid = (N/128, M/128), block = 256 (8 waves; wave = 32x64 sub-tile)
// ======================================================================
template<int MODE>
__global__ __launch_bounds__(256)
void gemm_bf16_wmma(const float* __restrict__ A, const float* __restrict__ Bm,
                    const float* __restrict__ bias,
                    const float* __restrict__ cosT, const float* __restrict__ sinT,
                    float* __restrict__ q_out, float* __restrict__ k_out,
                    float* __restrict__ v_out, float* __restrict__ C,
                    int N, int K, int ldb)
{
    const int LDSS = 34;                      // 32 K + 2 pad (keeps pairs aligned)
    __shared__ __bf16 sA[128 * 34];           // [row][k]
    __shared__ __bf16 sB[128 * 34];           // [col][k] (transposed stage)

    const int tid  = threadIdx.x;
    const int lane = tid & 31;
    const int wave = tid >> 5;
    const int wr   = wave & 3;                // row group: 32 rows
    const int wc   = wave >> 2;               // col group: 64 cols
    const int m0   = blockIdx.y * 128;
    const int n0   = blockIdx.x * 128;

    v8f acc[2][4];
#pragma unroll
    for (int rt = 0; rt < 2; ++rt)
#pragma unroll
        for (int ct = 0; ct < 4; ++ct)
#pragma unroll
            for (int e = 0; e < 8; ++e) acc[rt][ct][e] = 0.f;

    for (int k0 = 0; k0 < K; k0 += 32) {
        __syncthreads();
        {   // stage A tile 128x32: 2 threads/row, 16 elems each
            int row = tid >> 1, kb = (tid & 1) * 16;
            const float* gp = A + (size_t)(m0 + row) * K + k0 + kb;
#pragma unroll
            for (int i = 0; i < 16; i += 2)
                *(unsigned*)&sA[row * LDSS + kb + i] = pack_bf16(gp[i], gp[i + 1]);
        }
        {   // stage B tile 32x128 transposed: col-major into sB[col][k]
            int n = tid & 127, kb = (tid >> 7) * 16;
            const float* gp = Bm + (size_t)(k0 + kb) * ldb + n0 + n;
#pragma unroll
            for (int i = 0; i < 16; i += 2)
                *(unsigned*)&sB[n * LDSS + kb + i] =
                    pack_bf16(gp[(size_t)i * ldb], gp[(size_t)(i + 1) * ldb]);
        }
        __syncthreads();

        v16bf af[2], bf[4];
#pragma unroll
        for (int rt = 0; rt < 2; ++rt)
            af[rt] = load_frag_a<34>(sA, wr * 32 + rt * 16, 0, lane);
#pragma unroll
        for (int ct = 0; ct < 4; ++ct)
            bf[ct] = load_frag_b<34>(sB, wc * 64 + ct * 16, 0, lane);
#pragma unroll
        for (int rt = 0; rt < 2; ++rt)
#pragma unroll
            for (int ct = 0; ct < 4; ++ct)
                acc[rt][ct] = wmma_bf16(af[rt], bf[ct], acc[rt][ct]);
    }

    const int half = (lane >> 4) & 1;
    const int nn   = lane & 15;

    if (MODE == 0) {
#pragma unroll
        for (int rt = 0; rt < 2; ++rt)
#pragma unroll
            for (int ct = 0; ct < 4; ++ct) {
                int col = n0 + wc * 64 + ct * 16 + nn;
#pragma unroll
                for (int v = 0; v < 8; ++v) {
                    int row = m0 + wr * 32 + rt * 16 + v + 8 * half;
                    C[(size_t)row * N + col] = acc[rt][ct][v];
                }
            }
    } else {
        // n0 covers exactly one head of q/k/v
        const int sect = n0 / HID;                 // 0=q 1=k 2=v
        const int hoff = n0 % HID;                 // head*HS
        float* dst = (sect == 0) ? q_out : (sect == 1) ? k_out : v_out;
        float bv[4];
#pragma unroll
        for (int ct = 0; ct < 4; ++ct)
            bv[ct] = bias[n0 + wc * 64 + ct * 16 + nn];

        if (sect == 2 || wc == 1) {                // V, or dims 64..127: passthrough
#pragma unroll
            for (int rt = 0; rt < 2; ++rt)
#pragma unroll
                for (int ct = 0; ct < 4; ++ct) {
                    int col = wc * 64 + ct * 16 + nn;
#pragma unroll
                    for (int v = 0; v < 8; ++v) {
                        int t = m0 + wr * 32 + rt * 16 + v + 8 * half;
                        dst[(size_t)t * HID + hoff + col] = acc[rt][ct][v] + bv[ct];
                    }
                }
        } else {                                   // q/k dims 0..63: rotary pairs (ct, ct+2)
#pragma unroll
            for (int rt = 0; rt < 2; ++rt)
#pragma unroll
                for (int ct = 0; ct < 2; ++ct) {
                    int d = ct * 16 + nn;          // 0..31
#pragma unroll
                    for (int v = 0; v < 8; ++v) {
                        int t  = m0 + wr * 32 + rt * 16 + v + 8 * half;
                        float x1 = acc[rt][ct][v]     + bv[ct];
                        float x2 = acc[rt][ct + 2][v] + bv[ct + 2];
                        float c  = cosT[(size_t)t * ROT_HALF + d];
                        float s  = sinT[(size_t)t * ROT_HALF + d];
                        dst[(size_t)t * HID + hoff + d]      = x1 * c - x2 * s;
                        dst[(size_t)t * HID + hoff + d + 32] = x2 * c + x1 * s;
                    }
                }
        }
    }
}

// ======================================================================
// Causal flash attention. grid = (S/128, NH, B), block = 256 (8 waves).
// Wave owns 16 query rows; K-block in LDS [key][d]; V staged [d][key];
// P round-trips via per-wave LDS slice to re-enter WMMA A layout.
// ======================================================================
__device__ inline float red16_max(float x) {
    x = fmaxf(x, __shfl_xor(x, 1, 32));
    x = fmaxf(x, __shfl_xor(x, 2, 32));
    x = fmaxf(x, __shfl_xor(x, 4, 32));
    x = fmaxf(x, __shfl_xor(x, 8, 32));
    return x;
}
__device__ inline float red16_sum(float x) {
    x += __shfl_xor(x, 1, 32);
    x += __shfl_xor(x, 2, 32);
    x += __shfl_xor(x, 4, 32);
    x += __shfl_xor(x, 8, 32);
    return x;
}

__global__ __launch_bounds__(256)
void flash_attn(const float* __restrict__ Qr,   // (T, NH, HS) rotated q
                const float* __restrict__ Kc,   // (T, NH, HS) kv_k
                const float* __restrict__ Vc,   // (T, NH, HS) kv_v
                float* __restrict__ attn)       // (T, NH*HS)
{
    __shared__ __bf16 sKV[128 * 128];           // Q stage -> K blocks -> V^T blocks
    __shared__ __bf16 sP[8 * 16 * 128];         // per-wave P slices

    const int tid  = threadIdx.x;
    const int lane = tid & 31;
    const int wave = tid >> 5;
    const int qb   = blockIdx.x;
    const int h    = blockIdx.y;
    const int b    = blockIdx.z;
    const int t0   = b * S_LEN;
    const int qrow0 = qb * 128 + wave * 16;     // local (within batch seq)
    const int half = (lane >> 4) & 1;
    const int nn   = lane & 15;

    {   // stage 128x128 Q block as bf16 [row][d]
        int row = tid >> 1, db = (tid & 1) * 64;
        const float* gp = Qr + (size_t)(t0 + qb * 128 + row) * HID + h * HS + db;
#pragma unroll
        for (int i = 0; i < 64; i += 2)
            *(unsigned*)&sKV[row * 128 + db + i] = pack_bf16(gp[i], gp[i + 1]);
    }
    __syncthreads();
    v16bf qf[4];
#pragma unroll
    for (int kk = 0; kk < 4; ++kk)
        qf[kk] = load_frag_a<128>(sKV, wave * 16, kk * 32, lane);

    v8f o[8];
#pragma unroll
    for (int ct = 0; ct < 8; ++ct)
#pragma unroll
        for (int e = 0; e < 8; ++e) o[ct][e] = 0.f;
    float m_[8], l_[8];
#pragma unroll
    for (int v = 0; v < 8; ++v) { m_[v] = -1e30f; l_[v] = 0.f; }

    const float scale = 0.08838834764831845f;   // 1/sqrt(128)

    for (int kb = 0; kb <= qb; ++kb) {
        __syncthreads();                         // done with sKV contents
        {   // stage K block [key][d]
            int row = tid >> 1, db = (tid & 1) * 64;
            const float* gp = Kc + (size_t)(t0 + kb * 128 + row) * HID + h * HS + db;
#pragma unroll
            for (int i = 0; i < 64; i += 2)
                *(unsigned*)&sKV[row * 128 + db + i] = pack_bf16(gp[i], gp[i + 1]);
        }
        __syncthreads();

        // S = Q K^T (contract over d = 128)
        v8f s[8];
#pragma unroll
        for (int ct = 0; ct < 8; ++ct) {
#pragma unroll
            for (int e = 0; e < 8; ++e) s[ct][e] = 0.f;
#pragma unroll
            for (int kk = 0; kk < 4; ++kk) {
                v16bf bfrag = load_frag_b<128>(sKV, ct * 16, kk * 32, lane);
                s[ct] = wmma_bf16(qf[kk], bfrag, s[ct]);
            }
        }

        // scale + causal mask
        if (kb == qb) {
#pragma unroll
            for (int ct = 0; ct < 8; ++ct)
#pragma unroll
                for (int v = 0; v < 8; ++v) {
                    int qr = qrow0 + v + 8 * half;
                    int kc = kb * 128 + ct * 16 + nn;
                    s[ct][v] = (kc > qr) ? -1e30f : s[ct][v] * scale;
                }
        } else {
#pragma unroll
            for (int ct = 0; ct < 8; ++ct)
#pragma unroll
                for (int v = 0; v < 8; ++v) s[ct][v] *= scale;
        }

        // online softmax (row = v + 8*half, cols across tiles + 16 lanes)
#pragma unroll
        for (int v = 0; v < 8; ++v) {
            float mx = s[0][v];
#pragma unroll
            for (int ct = 1; ct < 8; ++ct) mx = fmaxf(mx, s[ct][v]);
            mx = red16_max(mx);
            float mnew = fmaxf(m_[v], mx);
            float corr = __expf(m_[v] - mnew);
            m_[v] = mnew;
            float rs = 0.f;
#pragma unroll
            for (int ct = 0; ct < 8; ++ct) {
                float p = __expf(s[ct][v] - mnew);
                s[ct][v] = p;
                rs += p;
            }
            rs = red16_sum(rs);
            l_[v] = l_[v] * corr + rs;
#pragma unroll
            for (int ct = 0; ct < 8; ++ct) o[ct][v] *= corr;
        }

        __syncthreads();                         // all waves done reading K
        {   // write P (bf16) to per-wave slice, A layout source [row][key]
            __bf16* myP = sP + wave * 16 * 128;
#pragma unroll
            for (int ct = 0; ct < 8; ++ct)
#pragma unroll
                for (int v = 0; v < 8; ++v)
                    myP[(v + 8 * half) * 128 + ct * 16 + nn] = (__bf16)s[ct][v];
        }
        {   // stage V block transposed: sKV[d][key]
            int key = tid >> 1, db = (tid & 1) * 64;
            const float* gp = Vc + (size_t)(t0 + kb * 128 + key) * HID + h * HS + db;
#pragma unroll
            for (int i = 0; i < 64; ++i)
                sKV[(db + i) * 128 + key] = (__bf16)gp[i];
        }
        __syncthreads();

        // O += P V (contract over key = 128)
        const __bf16* myP = sP + wave * 16 * 128;
#pragma unroll
        for (int kk = 0; kk < 4; ++kk) {
            v16bf pa = load_frag_a<128>(myP, 0, kk * 32, lane);
#pragma unroll
            for (int ct = 0; ct < 8; ++ct) {
                v16bf vb = load_frag_b<128>(sKV, ct * 16, kk * 32, lane);
                o[ct] = wmma_bf16(pa, vb, o[ct]);
            }
        }
    }

    // normalize + store: attn[t, h*HS + d]
#pragma unroll
    for (int v = 0; v < 8; ++v) {
        float inv = 1.f / l_[v];
        int t = t0 + qrow0 + v + 8 * half;
#pragma unroll
        for (int ct = 0; ct < 8; ++ct)
            attn[(size_t)t * HID + h * HS + ct * 16 + nn] = o[ct][v] * inv;
    }
}

// ======================================================================
extern "C" void kernel_launch(void* const* d_in, const int* in_sizes, int n_in,
                              void* d_out, int out_size, void* d_ws, size_t ws_size,
                              hipStream_t stream) {
    const float* hidden  = (const float*)d_in[0];
    const float* cosT    = (const float*)d_in[1];
    const float* sinT    = (const float*)d_in[2];
    const float* w_qkv   = (const float*)d_in[3];
    const float* b_qkv   = (const float*)d_in[4];
    const float* w_dense = (const float*)d_in[5];

    float* out  = (float*)d_out;
    float* kv_k = out + (size_t)T_TOK * HID;
    float* kv_v = kv_k + (size_t)T_TOK * HID;

    float* q_rot = (float*)d_ws;                       // 64 MB
    float* attn  = q_rot + (size_t)T_TOK * HID;        // 64 MB

    // 1) QKV GEMM + bias + RoPE + scatter (k/v land directly in output caches)
    gemm_bf16_wmma<1><<<dim3(3 * HID / 128, T_TOK / 128), 256, 0, stream>>>(
        hidden, w_qkv, b_qkv, cosT, sinT, q_rot, kv_k, kv_v, nullptr,
        3 * HID, HID, 3 * HID);

    // 2) causal flash attention
    flash_attn<<<dim3(S_LEN / 128, NH, B_SZ), 256, 0, stream>>>(
        q_rot, kv_k, kv_v, attn);

    // 3) dense projection
    gemm_bf16_wmma<0><<<dim3(HID / 128, T_TOK / 128), 256, 0, stream>>>(
        attn, w_dense, nullptr, nullptr, nullptr, nullptr, nullptr, nullptr, out,
        HID, HID, HID);
}
